// LearnerHead_34608846471216
// MI455X (gfx1250) — matrix operand order repaired and checked by reading, not compiled, not run on hardware
//
#include <hip/hip_runtime.h>
#include <hip/hip_bf16.h>

// Shapes from the reference
#define B_DIM  4
#define N_DIM  4096
#define D_IN   256
#define D_OUT  128

typedef __attribute__((ext_vector_type(16))) _Float16 v16h;
typedef __attribute__((ext_vector_type(8)))  _Float16 v8h;
typedef __attribute__((ext_vector_type(8)))  float    v8f;
typedef __attribute__((ext_vector_type(4)))  float    f32x4;
typedef __attribute__((ext_vector_type(4)))  int      i32x4;

#define LEAKY_SLOPE 0.01f
#define COS_EPS     0.001f
#define THRESHOLD   0.1f

#if __has_builtin(__builtin_amdgcn_global_load_async_to_lds_b128) && \
    __has_builtin(__builtin_amdgcn_s_wait_asynccnt)
#define USE_ASYNC_LDS 1
#else
#define USE_ASYNC_LDS 0
#endif

__device__ __forceinline__ float fast_rcp(float x) {
#if __has_builtin(__builtin_amdgcn_rcpf)
    return __builtin_amdgcn_rcpf(x);   // v_rcp_f32: ample for a 0.1 threshold
#else
    return 1.0f / x;
#endif
}

// ---------------------------------------------------------------------------
// Kernel 1: h = leaky_relu(Z @ W)  -> store h as f16 + exact f32 row norms.
// 8 rows per 128-thread block; Z rows staged in LDS, W rows broadcast from L2.
// ---------------------------------------------------------------------------
__global__ __launch_bounds__(128) void gemm1_norm_kernel(
    const float* __restrict__ Z, const float* __restrict__ W,
    _Float16* __restrict__ H, float* __restrict__ norms)
{
    __shared__ float zrow[8 * D_IN];      // 8 KB
    __shared__ float partial[8][4];

    const int tid   = threadIdx.x;        // 0..127, == output column d
    const int mbase = blockIdx.x * 8;

    // Stage 8 Z rows (2048 contiguous floats); Z is read exactly once -> NT
    const f32x4* zsrc = (const f32x4*)(Z + (size_t)mbase * D_IN);
    f32x4*       zdst = (f32x4*)zrow;
    #pragma unroll
    for (int i = tid; i < (8 * D_IN) / 4; i += 128)
        zdst[i] = __builtin_nontemporal_load(zsrc + i);
    __syncthreads();

    float acc[8] = {0.f, 0.f, 0.f, 0.f, 0.f, 0.f, 0.f, 0.f};
    #pragma unroll 4
    for (int k = 0; k < D_IN; ++k) {
        const float wv = W[k * D_OUT + tid];   // coalesced; W stays L2-hot
        #pragma unroll
        for (int r = 0; r < 8; ++r)
            acc[r] = fmaf(zrow[r * D_IN + k], wv, acc[r]);
    }

    const int lane = tid & 31;
    const int wid  = tid >> 5;
    #pragma unroll
    for (int r = 0; r < 8; ++r) {
        float v = acc[r];
        v = (v > 0.f) ? v : LEAKY_SLOPE * v;       // leaky_relu
        H[(size_t)(mbase + r) * D_OUT + tid] = (_Float16)v;
        float s = v * v;                            // wave32 reduction
        #pragma unroll
        for (int off = 16; off > 0; off >>= 1)
            s += __shfl_down(s, off, 32);
        if (lane == 0) partial[r][wid] = s;
    }
    __syncthreads();
    if (tid < 8)
        norms[mbase + tid] =
            sqrtf(partial[tid][0] + partial[tid][1] + partial[tid][2] + partial[tid][3]);
}

// ---------------------------------------------------------------------------
// Kernel 2: per batch, dots = h @ h^T via v_wmma_f32_16x16x32_f16, then
// abs(dot * rcp(max(n_i*n_j, eps))) with threshold.  128x64 tile / 256 thr.
// ---------------------------------------------------------------------------
__global__ __launch_bounds__(256) void cos_adj_kernel(
    const _Float16* __restrict__ H, const float* __restrict__ norms,
    float* __restrict__ out)
{
    __shared__ __align__(16) _Float16 Ai[128 * D_OUT];  // 32 KB
    __shared__ __align__(16) _Float16 Bj[64 * D_OUT];   // 16 KB
    __shared__ float nI[128];
    __shared__ float nJ[64];

    const int tid   = threadIdx.x;
    const int b     = blockIdx.z;
    const int ibase = blockIdx.y * 128;
    const int jbase = blockIdx.x * 64;

    const _Float16* Hb = H + (size_t)b * N_DIM * D_OUT;
    const i32x4* Hu = (const i32x4*)Hb;          // 8 halves per b128, 16/row
    i32x4* Au = (i32x4*)Ai;
    i32x4* Bu = (i32x4*)Bj;

#if USE_ASYNC_LDS
    // ASYNCcnt-tracked global->LDS copy: no VGPR round-trip for staging.
    #pragma unroll
    for (int i = tid; i < 128 * 16; i += 256) {  // A tile: 2048 x b128
        const int row = i >> 4, cw = i & 15;
        __builtin_amdgcn_global_load_async_to_lds_b128(
            (i32x4*)(Hu + (size_t)(ibase + row) * 16 + cw),
            (i32x4*)(Au + i), 0, 0);
    }
    #pragma unroll
    for (int i = tid; i < 64 * 16; i += 256) {   // B tile: 1024 x b128
        const int row = i >> 4, cw = i & 15;
        __builtin_amdgcn_global_load_async_to_lds_b128(
            (i32x4*)(Hu + (size_t)(jbase + row) * 16 + cw),
            (i32x4*)(Bu + i), 0, 0);
    }
    if (tid < 128) nI[tid] = norms[(size_t)b * N_DIM + ibase + tid];
    if (tid < 64)  nJ[tid] = norms[(size_t)b * N_DIM + jbase + tid];
    __builtin_amdgcn_s_wait_asynccnt(0);
#else
    #pragma unroll
    for (int i = tid; i < 128 * 16; i += 256) {  // A tile: 2048 x b128
        const int row = i >> 4, cw = i & 15;
        Au[i] = Hu[(size_t)(ibase + row) * 16 + cw];
    }
    #pragma unroll
    for (int i = tid; i < 64 * 16; i += 256) {   // B tile: 1024 x b128
        const int row = i >> 4, cw = i & 15;
        Bu[i] = Hu[(size_t)(jbase + row) * 16 + cw];
    }
    if (tid < 128) nI[tid] = norms[(size_t)b * N_DIM + ibase + tid];
    if (tid < 64)  nJ[tid] = norms[(size_t)b * N_DIM + jbase + tid];
#endif
    __syncthreads();

    const int lane  = tid & 31;
    const int w     = tid >> 5;          // wave id: owns rows [16w, 16w+16)
    const int lrow  = lane & 15;
    const int ahalf = (lane & 16) >> 1;  // A frag: K 0-7/16-23 vs 8-15/24-31
    const int bhalf = (lane & 16) >> 1;  // B frag: K 0-7/16-23 vs 8-15/24-31

    v8f c[4] = {};

    #pragma unroll
    for (int ks = 0; ks < 4; ++ks) {     // K = 128 -> 4 steps of 32
        const int kb = ks * 32;
        const _Float16* ap = &Ai[(16 * w + lrow) * D_OUT + kb + ahalf];
        const v8h alo = *(const v8h*)(ap);        // ds_load_b128
        const v8h ahi = *(const v8h*)(ap + 16);   // ds_load_b128
        v16h a;
        #pragma unroll
        for (int i = 0; i < 8; ++i) { a[i] = alo[i]; a[i + 8] = ahi[i]; }

        #pragma unroll
        for (int jt = 0; jt < 4; ++jt) {
            const _Float16* bp = &Bj[(16 * jt + lrow) * D_OUT + kb + bhalf];
            const v8h blo = *(const v8h*)(bp);
            const v8h bhi = *(const v8h*)(bp + 16);
            v16h bm;
            #pragma unroll
            for (int i = 0; i < 8; ++i) { bm[i] = blo[i]; bm[i + 8] = bhi[i]; }
            // D = A*B + C, f32 accumulate
            c[jt] = __builtin_amdgcn_wmma_f32_16x16x32_f16(
                false, a, false, bm, (short)0, c[jt], false, false);
        }
    }

    // Epilogue: C/D layout -> VGPR r holds M=r (lanes 0-15) / M=r+8 (16-31).
    // Output is written once and never re-read -> non-temporal stores keep
    // the 256 MB stream from evicting the L2-resident 4 MB h_f16.
    const int rowhalf = (lane & 16) >> 1;    // 0 or 8
    #pragma unroll
    for (int jt = 0; jt < 4; ++jt) {
        const int col_local = 16 * jt + lrow;
        const float nj = nJ[col_local];
        #pragma unroll
        for (int r = 0; r < 8; ++r) {
            const int row_local = 16 * w + r + rowhalf;
            const float denom = fmaxf(nI[row_local] * nj, COS_EPS);
            float v = fabsf(c[jt][r]) * fast_rcp(denom);
            v = (v > THRESHOLD) ? v : 0.0f;
            __builtin_nontemporal_store(
                v, out + ((size_t)b * N_DIM + ibase + row_local) * (size_t)N_DIM +
                        (size_t)(jbase + col_local));
        }
    }
}

// ---------------------------------------------------------------------------
extern "C" void kernel_launch(void* const* d_in, const int* in_sizes, int n_in,
                              void* d_out, int out_size, void* d_ws, size_t ws_size,
                              hipStream_t stream) {
    (void)in_sizes; (void)n_in; (void)out_size; (void)ws_size;
    const float* Z = (const float*)d_in[0];   // [4,4096,256] f32
    const float* W = (const float*)d_in[1];   // [256,128]    f32
    float* out = (float*)d_out;               // [4,4096,4096] f32

    // Workspace: h_f16 (4 MB) then norms (64 KB)
    _Float16* H     = (_Float16*)d_ws;
    float*    norms = (float*)((char*)d_ws + (size_t)B_DIM * N_DIM * D_OUT * sizeof(_Float16));

    // Kernel 1: 16384 rows / 8 rows per block
    gemm1_norm_kernel<<<(B_DIM * N_DIM) / 8, 128, 0, stream>>>(Z, W, H, norms);

    // Kernel 2: tiles of 128(i) x 64(j) per batch
    dim3 grid2(N_DIM / 64, N_DIM / 128, B_DIM);
    cos_adj_kernel<<<grid2, 256, 0, stream>>>(H, norms, out);
}
